// MambaBlock_12661563588755
// MI455X (gfx1250) — matrix operand order repaired and checked by reading, not compiled, not run on hardware
//
#include <hip/hip_runtime.h>
#include <hip/hip_bf16.h>
#include <stdint.h>

// ---------------- problem constants ----------------
#define BATCH    4
#define SEQ      2048
#define NTOK     (BATCH*SEQ)          // 8192
#define DMODEL   1024
#define DINNER   2048
#define DSTATE   128
#define NHEADS   32
#define HEADDIM  64
#define CHUNK    64
#define NCHUNK   (SEQ/CHUNK)          // 32
#define CONVDIM  2304                 // DINNER + 2*DSTATE
#define DPROJ    4384                 // 2*DINNER + 2*DSTATE + NHEADS
#define DPROJP   4480                 // padded to multiple of 128
#define EPSV     1e-5f

typedef __attribute__((ext_vector_type(2)))  float    f32x2;
typedef __attribute__((ext_vector_type(8)))  float    f32x8;
typedef __attribute__((ext_vector_type(4)))  int      v4i;
typedef __attribute__((ext_vector_type(4)))  unsigned u32x4;
typedef __attribute__((ext_vector_type(8)))  int      i32x8;
typedef __attribute__((ext_vector_type(8)))  __bf16   bf16x8;
typedef __attribute__((ext_vector_type(16))) __bf16   bf16x16;

union Frag16 { bf16x16 v; bf16x8 h[2]; };

#if defined(__has_builtin)
#if __has_builtin(__builtin_amdgcn_global_load_async_to_lds_b128)
#define HAVE_ASYNC_LDS 1
#endif
#if __has_builtin(__builtin_amdgcn_s_wait_asynccnt)
#define HAVE_WAIT_ASYNC 1
#endif
#if __has_builtin(__builtin_amdgcn_wmma_f32_16x16x4_f32)
#define HAVE_WMMA_F32 1
#endif
#if __has_builtin(__builtin_amdgcn_tensor_load_to_lds)
#define HAVE_TDM 1
#endif
#if __has_builtin(__builtin_amdgcn_s_wait_tensorcnt)
#define HAVE_WAIT_TENSOR 1
#endif
#endif

#define AS_GLOBAL __attribute__((address_space(1)))
#define AS_LOCAL  __attribute__((address_space(3)))

__device__ inline f32x8 f32x8_zero() {
  f32x8 z;
#pragma unroll
  for (int i = 0; i < 8; ++i) z[i] = 0.f;
  return z;
}

__device__ inline float silu_f(float x) { return x / (1.f + __expf(-x)); }

// 16-byte global -> LDS copy (async engine on CDNA5, sync fallback otherwise)
__device__ inline void copy16_g2l(const __bf16* g, __bf16* l) {
#if defined(HAVE_ASYNC_LDS)
  // AS1 pointer value == flat address; AS3 offset == low 32 bits of flat LDS addr
  AS_GLOBAL v4i* gp = (AS_GLOBAL v4i*)(unsigned long long)(uintptr_t)g;
  AS_LOCAL  v4i* lp = (AS_LOCAL  v4i*)(unsigned)(uintptr_t)l;
  __builtin_amdgcn_global_load_async_to_lds_b128(gp, lp, 0, 0);
#else
  *(bf16x8*)l = *(const bf16x8*)g;
#endif
}

__device__ inline void async_fence() {
#if defined(HAVE_ASYNC_LDS)
#if defined(HAVE_WAIT_ASYNC)
  __builtin_amdgcn_s_wait_asynccnt(0);
#else
  asm volatile("s_wait_asynccnt 0x0" ::: "memory");
#endif
#endif
}

__device__ inline void tensor_fence() {
#if defined(HAVE_TDM)
#if defined(HAVE_WAIT_TENSOR)
  __builtin_amdgcn_s_wait_tensorcnt(0);
#else
  asm volatile("s_wait_tensorcnt 0x0" ::: "memory");
#endif
#endif
}

#if defined(HAVE_TDM)
// TDM: DMA a 2D bf16 tile (tile_rows x 32 elems, row stride = stride_e elems)
// into LDS with HW padding of 4 dwords after every 16 dwords -> 40-half rows.
__device__ inline void tdm_load_tile_bf16(const __bf16* gsrc, unsigned lds_off,
                                          int stride_e, int tile_rows) {
  const unsigned long long ga = (unsigned long long)(uintptr_t)gsrc;
  u32x4 g0;
  g0[0] = 1u;                                          // count=1 (valid D#)
  g0[1] = lds_off;                                     // lds_addr (bytes)
  g0[2] = (unsigned)(ga & 0xffffffffull);              // global_addr[31:0]
  g0[3] = (unsigned)((ga >> 32) & 0x1ffffffull)        // global_addr[56:32]
        | (2u << 30);                                  // type=2 ("image")
  i32x8 g1;
  g1[0] = (int)((1u << 16)                             // data_size = 2 bytes
              | (1u << 20)                             // pad_enable
              | (3u << 22)                             // pad_interval: 16 dwords
              | (3u << 25));                           // pad_amount: 4 dwords
  g1[1] = (int)(((unsigned)stride_e & 0xffffu) << 16); // tensor_dim0[15:0]
  g1[2] = (int)((((unsigned)stride_e >> 16) & 0xffffu) // tensor_dim0[31:16]
              | (((unsigned)tile_rows & 0xffffu) << 16)); // tensor_dim1[15:0]
  g1[3] = (int)(32u << 16);                            // tensor_dim1 hi=0 | tile_dim0=32
  g1[4] = (int)((unsigned)tile_rows & 0xffffu);        // tile_dim1 | tile_dim2=0
  g1[5] = stride_e;                                    // tensor_dim0_stride[31:0]
  g1[6] = 0;                                           // stride0 hi | stride1 lo
  g1[7] = 0;
  v4i gz = {0, 0, 0, 0};
#if __clang_major__ >= 23
  i32x8 gz8 = {0, 0, 0, 0, 0, 0, 0, 0};
  __builtin_amdgcn_tensor_load_to_lds(g0, g1, gz, gz, gz8, 0);
#else
  __builtin_amdgcn_tensor_load_to_lds(g0, g1, gz, gz, 0);
#endif
}
#endif

// ---------------- 1. LayerNorm + bf16 cast ----------------
__global__ __launch_bounds__(256)
void mamba_ln_kernel(const float* __restrict__ x, const float* __restrict__ g,
                     const float* __restrict__ b, __bf16* __restrict__ out) {
  __shared__ float red[256];
  const int t = blockIdx.x, tid = threadIdx.x;
  const float* row = x + (size_t)t * DMODEL;
  float4 v = ((const float4*)row)[tid];
  float s = v.x + v.y + v.z + v.w;
  red[tid] = s; __syncthreads();
  for (int o = 128; o > 0; o >>= 1) { if (tid < o) red[tid] += red[tid + o]; __syncthreads(); }
  const float mu = red[0] * (1.f / DMODEL);
  __syncthreads();
  float d0 = v.x - mu, d1 = v.y - mu, d2 = v.z - mu, d3 = v.w - mu;
  red[tid] = d0*d0 + d1*d1 + d2*d2 + d3*d3; __syncthreads();
  for (int o = 128; o > 0; o >>= 1) { if (tid < o) red[tid] += red[tid + o]; __syncthreads(); }
  const float rinv = rsqrtf(red[0] * (1.f / DMODEL) + EPSV);
  const int c = tid * 4;
  __bf16* orow = out + (size_t)t * DMODEL + c;
  orow[0] = (__bf16)(d0 * rinv * g[c+0] + b[c+0]);
  orow[1] = (__bf16)(d1 * rinv * g[c+1] + b[c+1]);
  orow[2] = (__bf16)(d2 * rinv * g[c+2] + b[c+2]);
  orow[3] = (__bf16)(d3 * rinv * g[c+3] + b[c+3]);
}

// ---------------- 2. weight casts ----------------
__global__ void mamba_cast_win_kernel(const float* __restrict__ w, __bf16* __restrict__ o) {
  const int n = DPROJP * DMODEL;
  for (int i = blockIdx.x * blockDim.x + threadIdx.x; i < n; i += gridDim.x * blockDim.x) {
    int r = i / DMODEL;
    o[i] = (__bf16)(r < DPROJ ? w[(size_t)r * DMODEL + (i - r * DMODEL)] : 0.f);
  }
}
__global__ void mamba_cast_kernel(const float* __restrict__ w, __bf16* __restrict__ o, int n) {
  for (int i = blockIdx.x * blockDim.x + threadIdx.x; i < n; i += gridDim.x * blockDim.x)
    o[i] = (__bf16)w[i];
}

// ---------------- 3. WMMA bf16 GEMM: C[M,N] = A[M,K] * B[N,K]^T (+resid) ----------------
#define BM 128
#define BN 128
#define BK 32
#define LDA 40
#define LDB 40

__global__ __launch_bounds__(256)
void mamba_gemm_kernel(const __bf16* __restrict__ A, const __bf16* __restrict__ Bw,
                       float* __restrict__ C, const float* __restrict__ resid,
                       int M, int N, int K) {
  __shared__ __bf16 As[2][BM * LDA];
  __shared__ __bf16 Bs[2][BN * LDB];
  const int tid = threadIdx.x;
  const int bm = blockIdx.y, bn = blockIdx.x;
  const int w = tid >> 5, l = tid & 31;
  const int wr = w >> 2, wc = w & 3;          // wave tile: 64 rows x 32 cols

  f32x8 acc[4][2];
#pragma unroll
  for (int i = 0; i < 4; ++i)
#pragma unroll
    for (int j = 0; j < 2; ++j) acc[i][j] = f32x8_zero();

  const int krow = tid >> 2;                  // 0..63
  const int kcol = (tid & 3) * 8;             // 0,8,16,24 halfs
  const int arow = wr * 64 + (l & 15);
  const int klo  = (l >> 4) * 8;              // A frag chunk base (halfs)
  const int kb   = (l >> 4) * 8;              // B frag chunk base (halfs)
  const int KT = K / BK;

  auto stage = [&](int buf, int k0) {
    // A tile: per-lane async gather (ASYNCcnt path)
#pragma unroll
    for (int p = 0; p < 2; ++p) {
      const int r = p * 64 + krow;
      copy16_g2l(A + (size_t)(bm * BM + r) * K + k0 + kcol, &As[buf][r * LDA + kcol]);
    }
    // B tile: one TDM 2D descriptor per workgroup (TENSORcnt path)
#if defined(HAVE_TDM)
    if (w == 0)
      tdm_load_tile_bf16(Bw + (size_t)(bn * BN) * K + k0,
                         (unsigned)(uintptr_t)&Bs[buf][0], K, BN);
#else
#pragma unroll
    for (int p = 0; p < 2; ++p) {
      const int r = p * 64 + krow;
      copy16_g2l(Bw + (size_t)(bn * BN + r) * K + k0 + kcol, &Bs[buf][r * LDB + kcol]);
    }
#endif
  };
  auto fence = [&]() {
    async_fence();     // this wave's async A-tile copies
    tensor_fence();    // nop for waves with TENSORcnt==0
  };

  stage(0, 0);
  fence();
  __syncthreads();

  for (int kt = 0; kt < KT; ++kt) {
    const int cur = kt & 1;
    if (kt + 1 < KT) stage(cur ^ 1, (kt + 1) * BK);
    if (kt + 2 < KT) {   // global_prefetch_b8 on gfx1250
      __builtin_prefetch(A + (size_t)(bm * BM + krow) * K + (kt + 2) * BK + kcol, 0, 1);
      __builtin_prefetch(Bw + (size_t)(bn * BN + krow) * K + (kt + 2) * BK + kcol, 0, 1);
    }

    Frag16 afr[4];
#pragma unroll
    for (int i = 0; i < 4; ++i) {
      const int rr = arow + i * 16;
      afr[i].h[0] = *(const bf16x8*)&As[cur][rr * LDA + klo];
      afr[i].h[1] = *(const bf16x8*)&As[cur][rr * LDA + klo + 16];
    }
    Frag16 bfr[2];
#pragma unroll
    for (int j = 0; j < 2; ++j) {
      const int cc = wc * 32 + j * 16 + (l & 15);
      bfr[j].h[0] = *(const bf16x8*)&Bs[cur][cc * LDB + kb];
      bfr[j].h[1] = *(const bf16x8*)&Bs[cur][cc * LDB + kb + 16];
    }
#pragma unroll
    for (int i = 0; i < 4; ++i)
#pragma unroll
      for (int j = 0; j < 2; ++j)
        acc[i][j] = __builtin_amdgcn_wmma_f32_16x16x32_bf16(
            false, afr[i].v, false, bfr[j].v, (short)0, acc[i][j], false, false);

    if (kt + 1 < KT) fence();
    __syncthreads();
  }

  // store: D layout -> lane = n + 16*(m>=8), vgpr r = m&7
  const int n = l & 15;
  const int mo = (l >> 4) * 8;
  const size_t mbase = (size_t)bm * BM + wr * 64;
  const int nbase = bn * BN + wc * 32;
#pragma unroll
  for (int i = 0; i < 4; ++i)
#pragma unroll
    for (int j = 0; j < 2; ++j) {
      const size_t rb = mbase + i * 16 + mo;
      const int cb = nbase + j * 16 + n;
#pragma unroll
      for (int r = 0; r < 8; ++r) {
        const size_t idx = (rb + r) * (size_t)N + cb;
        float vv = acc[i][j][r];
        if (resid) vv += resid[idx];
        C[idx] = vv;
      }
    }
}

// ---------------- 4. depthwise conv(4) + SiLU, softplus(dt) ----------------
__global__ __launch_bounds__(256)
void mamba_conv_kernel(const float* __restrict__ zx, const float* __restrict__ cw,
                       const float* __restrict__ cb, const float* __restrict__ dtb,
                       float* __restrict__ xbc, float* __restrict__ dtv) {
  const int t = blockIdx.x;
  const int lpos = t & (SEQ - 1);
  for (int ch = threadIdx.x; ch < CONVDIM; ch += 256) {
    float acc = cb[ch];
#pragma unroll
    for (int k = 0; k < 4; ++k) {
      const int lk = lpos - 3 + k;
      if (lk >= 0)
        acc += zx[(size_t)(t - 3 + k) * DPROJP + DINNER + ch] * cw[ch * 4 + k];
    }
    xbc[(size_t)t * CONVDIM + ch] = silu_f(acc);
  }
  for (int h = threadIdx.x; h < NHEADS; h += 256) {
    const float dr = zx[(size_t)t * DPROJP + DINNER + CONVDIM + h] + dtb[h];
    dtv[t * NHEADS + h] = dr > 20.f ? dr : log1pf(__expf(dr));
  }
}

// ---------------- 5. SSD chunk scan, one block per (batch, head) ----------------
#define NB 132   // padded DSTATE stride
#define NX 68    // padded HEADDIM/CHUNK stride

__global__ __launch_bounds__(256)
void mamba_ssd_kernel(const float* __restrict__ xbc, const float* __restrict__ dtv,
                      const float* __restrict__ A_log, const float* __restrict__ Dskip,
                      float* __restrict__ y) {
  extern __shared__ float sm[];
  float* Bc   = sm;                  // 64 x NB
  float* Cc   = Bc   + CHUNK * NB;   // 64 x NB
  float* prev = Cc   + CHUNK * NB;   // 64(p) x NB(n)
  float* xdt  = prev + HEADDIM * NB; // 64 x NX
  float* xsv  = xdt  + CHUNK * NX;   // 64 x NX
  float* G    = xsv  + CHUNK * NX;   // 64 x NX
  float* Yo   = G    + CHUNK * NX;   // 64 x NX
  float* acs  = Yo   + CHUNK * NX;   // 64
  float* wdec = acs  + CHUNK;        // 64

  const int bh = blockIdx.x, b = bh >> 5, h = bh & 31;
  const int tid = threadIdx.x;
  const int w = tid >> 5, l = tid & 31;
  const float Ah = -__expf(A_log[h]);
  const float Dh = Dskip[h];

  for (int i = tid; i < HEADDIM * NB; i += 256) prev[i] = 0.f;
  __syncthreads();

  for (int c = 0; c < NCHUNK; ++c) {
    const int t0 = b * SEQ + c * CHUNK;
    for (int i = tid; i < CHUNK * DSTATE; i += 256) {
      const int lr = i >> 7, n = i & 127;
      const float* r = xbc + (size_t)(t0 + lr) * CONVDIM;
      Bc[lr * NB + n] = r[DINNER + n];
      Cc[lr * NB + n] = r[DINNER + DSTATE + n];
    }
    for (int i = tid; i < CHUNK * HEADDIM; i += 256) {
      const int lr = i >> 6, p = i & 63;
      const float xv = xbc[(size_t)(t0 + lr) * CONVDIM + h * HEADDIM + p];
      const float dt = dtv[(t0 + lr) * NHEADS + h];
      xsv[lr * NX + p] = xv;
      xdt[lr * NX + p] = xv * dt;
    }
    if (tid == 0) {
      float s = 0.f;
      for (int lr = 0; lr < CHUNK; ++lr) { s += dtv[(t0 + lr) * NHEADS + h] * Ah; acs[lr] = s; }
      for (int lr = 0; lr < CHUNK; ++lr) wdec[lr] = __expf(acs[CHUNK - 1] - acs[lr]);
    }
    __syncthreads();

#if defined(HAVE_WMMA_F32)
    // ---- phase 2 (fp32 WMMA): G = (C B^T) .* Lmat  and  Yo = C prev^T ----
    {
      const int m = l & 15, mo = (l >> 4) * 8, nn = l & 15;
      const int kq = (l >> 4) * 2;  // f32 A/B frag: lane holds K = k4+kq, k4+kq+1
      for (int tt = w; tt < 16; tt += 8) {           // G: 4x4 tiles of 16x16
        const int l0 = (tt >> 2) * 16, s0 = (tt & 3) * 16;
        f32x8 acc = f32x8_zero();
        const float* ap = &Cc[(l0 + m) * NB + kq];
        const float* bp = &Bc[(s0 + m) * NB + kq];
#pragma unroll
        for (int k4 = 0; k4 < DSTATE; k4 += 4) {
          f32x2 av = { ap[k4], ap[k4 + 1] };
          f32x2 bv = { bp[k4], bp[k4 + 1] };
          acc = __builtin_amdgcn_wmma_f32_16x16x4_f32(
              false, av, false, bv, (short)0, acc, false, false);
        }
#pragma unroll
        for (int r = 0; r < 8; ++r) {
          const int lr = l0 + mo + r, s = s0 + nn;
          G[lr * NX + s] = (s <= lr) ? acc[r] * __expf(acs[lr] - acs[s]) : 0.f;
        }
      }
      for (int tt = w; tt < 16; tt += 8) {           // Yo: 4(l) x 4(p) tiles
        const int l0 = (tt >> 2) * 16, p0 = (tt & 3) * 16;
        f32x8 acc = f32x8_zero();
        const float* ap = &Cc[(l0 + m) * NB + kq];
        const float* bp = &prev[(p0 + m) * NB + kq];
#pragma unroll
        for (int k4 = 0; k4 < DSTATE; k4 += 4) {
          f32x2 av = { ap[k4], ap[k4 + 1] };
          f32x2 bv = { bp[k4], bp[k4 + 1] };
          acc = __builtin_amdgcn_wmma_f32_16x16x4_f32(
              false, av, false, bv, (short)0, acc, false, false);
        }
#pragma unroll
        for (int r = 0; r < 8; ++r)
          Yo[(l0 + mo + r) * NX + p0 + nn] = acc[r];
      }
    }
#else
    for (int i = tid; i < CHUNK * CHUNK; i += 256) {
      const int lr = i >> 6, s = i & 63;
      float gsum = 0.f;
      if (s <= lr) {
        const float* cp = &Cc[lr * NB];
        const float* bp = &Bc[s * NB];
        for (int n = 0; n < DSTATE; ++n) gsum += cp[n] * bp[n];
        gsum *= __expf(acs[lr] - acs[s]);
      }
      G[lr * NX + s] = gsum;
    }
    for (int i = tid; i < CHUNK * HEADDIM; i += 256) {
      const int lr = i >> 6, p = i & 63;
      float yo = 0.f;
      const float* cp = &Cc[lr * NB];
      for (int n = 0; n < DSTATE; ++n) yo += cp[n] * prev[p * NB + n];
      Yo[lr * NX + p] = yo;
    }
#endif
    __syncthreads();

    // ---- phase 3 (scalar): y = Y_diag + exp(acs)*Yo + D*xs ----
    for (int i = tid; i < CHUNK * HEADDIM; i += 256) {
      const int lr = i >> 6, p = i & 63;
      float yv = 0.f;
      for (int s = 0; s <= lr; ++s) yv += G[lr * NX + s] * xdt[s * NX + p];
      yv += __expf(acs[lr]) * Yo[lr * NX + p];
      yv += Dh * xsv[lr * NX + p];
      y[(size_t)(t0 + lr) * DINNER + h * HEADDIM + p] = yv;
    }

    // ---- phase 4: prev[p][n] = exp(acs[63]) * prev + sum_l wdec[l]*xdt[l,p]*B[l,n] ----
    const float eAl = __expf(acs[CHUNK - 1]);
#if defined(HAVE_WMMA_F32)
    {
      const int m = l & 15, mo = (l >> 4) * 8, nn = l & 15;
      const int kq = (l >> 4) * 2;
      for (int tt = w; tt < 32; tt += 8) {           // 4(p) x 8(n) tiles
        const int p0 = (tt >> 3) * 16, n0 = (tt & 7) * 16;
        f32x8 acc = f32x8_zero();
#pragma unroll
        for (int k4 = 0; k4 < CHUNK; k4 += 4) {
          const int kk = k4 + kq;
          f32x2 av = { xdt[kk * NX + p0 + m] * wdec[kk],
                       xdt[(kk + 1) * NX + p0 + m] * wdec[kk + 1] };
          f32x2 bv = { Bc[kk * NB + n0 + m], Bc[(kk + 1) * NB + n0 + m] };
          acc = __builtin_amdgcn_wmma_f32_16x16x4_f32(
              false, av, false, bv, (short)0, acc, false, false);
        }
#pragma unroll
        for (int r = 0; r < 8; ++r) {
          const int p = p0 + mo + r, n = n0 + nn;
          prev[p * NB + n] = eAl * prev[p * NB + n] + acc[r];
        }
      }
    }
#else
    for (int i = tid; i < HEADDIM * DSTATE; i += 256) {
      const int p = i >> 7, n = i & 127;
      float sa = 0.f;
      for (int lr = 0; lr < CHUNK; ++lr) sa += Bc[lr * NB + n] * wdec[lr] * xdt[lr * NX + p];
      prev[p * NB + n] = eAl * prev[p * NB + n] + sa;
    }
#endif
    __syncthreads();
  }
}

// ---------------- 6. SiLU gate + RMSNorm -> bf16 ----------------
__global__ __launch_bounds__(256)
void mamba_gate_kernel(const float* __restrict__ y, const float* __restrict__ zx,
                       const float* __restrict__ nw, __bf16* __restrict__ out) {
  __shared__ float red[256];
  const int t = blockIdx.x, tid = threadIdx.x;
  const float* yr = y + (size_t)t * DINNER;
  const float* zr = zx + (size_t)t * DPROJP;
  float vals[8]; float ss = 0.f;
#pragma unroll
  for (int i = 0; i < 8; ++i) {
    const int j = i * 256 + tid;
    const float v = yr[j] * silu_f(zr[j]);
    vals[i] = v; ss += v * v;
  }
  red[tid] = ss; __syncthreads();
  for (int o = 128; o > 0; o >>= 1) { if (tid < o) red[tid] += red[tid + o]; __syncthreads(); }
  const float rinv = rsqrtf(red[0] * (1.f / DINNER) + EPSV);
#pragma unroll
  for (int i = 0; i < 8; ++i) {
    const int j = i * 256 + tid;
    out[(size_t)t * DINNER + j] = (__bf16)(vals[i] * rinv * nw[j]);
  }
}

// ---------------- launch ----------------
extern "C" void kernel_launch(void* const* d_in, const int* in_sizes, int n_in,
                              void* d_out, int out_size, void* d_ws, size_t ws_size,
                              hipStream_t stream) {
  const float* x      = (const float*)d_in[0];
  const float* W_in   = (const float*)d_in[1];
  const float* conv_w = (const float*)d_in[2];
  const float* conv_b = (const float*)d_in[3];
  const float* dt_b   = (const float*)d_in[4];
  const float* A_log  = (const float*)d_in[5];
  const float* Dskip  = (const float*)d_in[6];
  const float* norm_w = (const float*)d_in[7];
  const float* W_out  = (const float*)d_in[8];
  const float* ln_g   = (const float*)d_in[9];
  const float* ln_b   = (const float*)d_in[10];
  float* out = (float*)d_out;

  char* ws = (char*)d_ws;
  size_t off = 0;
  auto take = [&](size_t bytes) { size_t r = off; off += (bytes + 255) & ~(size_t)255; return r; };
  __bf16* xn    = (__bf16*)(ws + take((size_t)NTOK * DMODEL * 2));
  __bf16* winb  = (__bf16*)(ws + take((size_t)DPROJP * DMODEL * 2));
  __bf16* woutb = (__bf16*)(ws + take((size_t)DMODEL * DINNER * 2));
  float*  zx    = (float*) (ws + take((size_t)NTOK * DPROJP * 4));
  float*  xbc   = (float*) (ws + take((size_t)NTOK * CONVDIM * 4));
  float*  dtv   = (float*) (ws + take((size_t)NTOK * NHEADS * 4));
  float*  yssd  = (float*) (ws + take((size_t)NTOK * DINNER * 4));
  __bf16* ygn   = (__bf16*)(ws + take((size_t)NTOK * DINNER * 2));

  const size_t ssd_smem =
      (3u * CHUNK * NB + 4u * CHUNK * NX + 2u * CHUNK) * sizeof(float);
  (void)hipFuncSetAttribute((const void*)mamba_ssd_kernel,
                            hipFuncAttributeMaxDynamicSharedMemorySize, (int)ssd_smem);

  mamba_ln_kernel<<<NTOK, 256, 0, stream>>>(x, ln_g, ln_b, xn);
  mamba_cast_win_kernel<<<2048, 256, 0, stream>>>(W_in, winb);
  mamba_cast_kernel<<<1024, 256, 0, stream>>>(W_out, woutb, DMODEL * DINNER);

  mamba_gemm_kernel<<<dim3(DPROJP / BN, NTOK / BM), 256, 0, stream>>>(
      xn, winb, zx, nullptr, NTOK, DPROJP, DMODEL);

  mamba_conv_kernel<<<NTOK, 256, 0, stream>>>(zx, conv_w, conv_b, dt_b, xbc, dtv);

  mamba_ssd_kernel<<<BATCH * NHEADS, 256, ssd_smem, stream>>>(xbc, dtv, A_log, Dskip, yssd);

  mamba_gate_kernel<<<NTOK, 256, 0, stream>>>(yssd, zx, norm_w, ygn);

  mamba_gemm_kernel<<<dim3(DMODEL / BN, NTOK / BM), 256, 0, stream>>>(
      ygn, woutb, out, x, NTOK, DMODEL, DINNER);
}
